// Attention_39814346834097
// MI455X (gfx1250) — matrix operand order repaired and checked by reading, not compiled, not run on hardware
//
#include <hip/hip_runtime.h>
#include <cstdint>
#include <cstddef>

// ---------------------------------------------------------------------------
// Types for CDNA5 WMMA (wave32): D(16x16 f32) = A(16x32 bf16) x B(32x16 bf16) + C
// ---------------------------------------------------------------------------
typedef __attribute__((ext_vector_type(16))) __bf16 v16bf;
typedef __attribute__((ext_vector_type(8)))  float  v8f;

union Frag16 { v16bf v; uint4 u[2]; };

__device__ __forceinline__ v8f wmma_bf16(v16bf a, v16bf b, v8f c) {
  // (neg_a, A, neg_b, B, c_mod, C, reuse_a, reuse_b)
  return __builtin_amdgcn_wmma_f32_16x16x32_bf16(false, a, false, b, (short)0, c, false, false);
}

// float -> bf16, round-to-nearest-even, via bit manipulation
__device__ __forceinline__ __bf16 f2bf(float f) {
  union { float f; uint32_t u; } a; a.f = f;
  uint32_t r = a.u + 0x7FFFu + ((a.u >> 16) & 1u);
  union { unsigned short s; __bf16 h; } o;
  o.s = (unsigned short)(r >> 16);
  return o.h;
}

// A fragment (16x32 bf16): lane l (m = l&15, hi = l>>4):
//   e[0..7]  = row[koff + hi*8 + t],  e[8..15] = row[koff + hi*8 + 16 + t]
__device__ __forceinline__ v16bf load_a_frag(const __bf16* __restrict__ row, int koff, int hi) {
  Frag16 f;
  const __bf16* p = row + koff + hi * 8;
  f.u[0] = *reinterpret_cast<const uint4*>(p);
  f.u[1] = *reinterpret_cast<const uint4*>(p + 16);
  return f.v;
}

// B fragment (32x16 bf16), column n = l&15 supplied as a contiguous "row" of B^T:
//   e[0..7]  = colrow[koff + hi*8 + t],  e[8..15] = colrow[koff + hi*8 + 16 + t]
__device__ __forceinline__ v16bf load_b_frag(const __bf16* __restrict__ colrow, int koff, int hi) {
  Frag16 f;
  const __bf16* p = colrow + koff + hi * 8;
  f.u[0] = *reinterpret_cast<const uint4*>(p);
  f.u[1] = *reinterpret_cast<const uint4*>(p + 16);
  return f.v;
}

// ---------------------------------------------------------------------------
// Problem constants
// ---------------------------------------------------------------------------
#define BATCH 4
#define SEQ   2048
#define DMODEL 1024
#define NHEAD 16
#define DHEAD 64
#define ROWS  (BATCH * SEQ)          // 8192
#define N_QKV (3 * DMODEL)           // 3072
#define SROW  2052                   // padded LDS row stride (floats); 8*2052 % 64 = 32 -> no bank conflict

// ---------------------------------------------------------------------------
// 1) fp32 -> bf16 cast (x), vectorized 4-wide
// ---------------------------------------------------------------------------
__global__ __launch_bounds__(256) void cast4_kernel(const float* __restrict__ in,
                                                    __bf16* __restrict__ out, int n4) {
  int i = blockIdx.x * blockDim.x + threadIdx.x;
  if (i >= n4) return;
  float4 f = reinterpret_cast<const float4*>(in)[i];
  union { __bf16 h[4]; ushort4 u; } o;
  o.h[0] = f2bf(f.x); o.h[1] = f2bf(f.y); o.h[2] = f2bf(f.z); o.h[3] = f2bf(f.w);
  reinterpret_cast<ushort4*>(out)[i] = o.u;
}

// ---------------------------------------------------------------------------
// 2) fp32 [rows][cols] -> bf16 transposed [cols][rows]
// ---------------------------------------------------------------------------
__global__ __launch_bounds__(256) void cast_tr_kernel(const float* __restrict__ w,
                                                      __bf16* __restrict__ wT,
                                                      int rows, int cols) {
  int i = blockIdx.x * blockDim.x + threadIdx.x;
  if (i >= rows * cols) return;
  int r = i / rows;   // column of w
  int c = i % rows;   // row of w
  wT[i] = f2bf(w[(size_t)c * cols + r]);
}

// ---------------------------------------------------------------------------
// 3) QKV GEMM: [8192 x 1024] x [1024 x 3072]; wave tile = 32x64
//    k-step: batch-load 2 A + 4 B fragments (distinct regs -> one load clause),
//    then 8 WMMA drain with graduated waitcnts.
//    scatter: q(scaled)/k -> [b,h,n,d], v -> [b,h,d,n] (transposed)
// ---------------------------------------------------------------------------
__global__ __launch_bounds__(256) void qkv_gemm_kernel(const __bf16* __restrict__ xb,
                                                       const __bf16* __restrict__ wT,
                                                       __bf16* __restrict__ q,
                                                       __bf16* __restrict__ k,
                                                       __bf16* __restrict__ vT) {
  int wave = blockIdx.x * (blockDim.x >> 5) + (threadIdx.x >> 5);
  int lane = threadIdx.x & 31, l15 = lane & 15, hi = lane >> 4;
  const int MT = ROWS / 32;                    // 256
  int tm = wave % MT;
  int tn = wave / MT;                          // 0..47
  const __bf16* a0 = xb + (size_t)(tm * 32 + l15) * DMODEL;
  const __bf16* a1 = a0 + 16 * DMODEL;
  const __bf16* b0 = wT + (size_t)(tn * 64 + l15) * DMODEL;

  v8f acc[2][4] = {};
  for (int kk = 0; kk < DMODEL; kk += 32) {
    __builtin_prefetch(a0 + kk + 256, 0, 3);
    __builtin_prefetch(b0 + kk + 256, 0, 3);
    v16bf fa0 = load_a_frag(a0, kk, hi);
    v16bf fa1 = load_a_frag(a1, kk, hi);
    v16bf fb[4];
#pragma unroll
    for (int j = 0; j < 4; ++j)
      fb[j] = load_b_frag(b0 + (size_t)j * 16 * DMODEL, kk, hi);
#pragma unroll
    for (int j = 0; j < 4; ++j) {
      acc[0][j] = wmma_bf16(fa0, fb[j], acc[0][j]);
      acc[1][j] = wmma_bf16(fa1, fb[j], acc[1][j]);
    }
  }

  // Epilogue: C layout: lane l, vgpr r -> M = r + 8*hi, N = l&15
#pragma unroll
  for (int i = 0; i < 2; ++i) {
#pragma unroll
    for (int j = 0; j < 4; ++j) {
      int N = tn * 64 + j * 16 + l15;
#pragma unroll
      for (int r = 0; r < 8; ++r) {
        int M = tm * 32 + i * 16 + r + 8 * hi;
        int b = M >> 11, n = M & (SEQ - 1);
        float val = acc[i][j][r];
        if (N < DMODEL) {                              // Q, pre-scaled by 1/sqrt(64)
          int h = N >> 6, d = N & 63;
          q[(((size_t)(b * NHEAD + h) * SEQ + n) << 6) + d] = f2bf(val * 0.125f);
        } else if (N < 2 * DMODEL) {                   // K
          int N2 = N - DMODEL; int h = N2 >> 6, d = N2 & 63;
          k[(((size_t)(b * NHEAD + h) * SEQ + n) << 6) + d] = f2bf(val);
        } else {                                       // V, transposed [b,h,d,n]
          int N2 = N - 2 * DMODEL; int h = N2 >> 6, d = N2 & 63;
          vT[(((size_t)(b * NHEAD + h) * DHEAD + d) << 11) + n] = f2bf(val);
        }
      }
    }
  }
}

// ---------------------------------------------------------------------------
// 4) Attention: one workgroup (8 waves) per 32 query rows per (b,h).
//    S = QK^T (WMMA) -> LDS fp32 [32][SROW]
//    softmax: max-reduce, then fused exp + bf16-pack (each thread owns an
//    exclusive 256-col chunk; bf16 overlays the front of its own chunk),
//    1/sum deferred to the PV epilogue.
//    O = P V (WMMA; A read directly as bf16 from LDS), -> bf16 [b, n, h*64+d].
// ---------------------------------------------------------------------------
__global__ __launch_bounds__(256) void attn_kernel(const __bf16* __restrict__ q,
                                                   const __bf16* __restrict__ k,
                                                   const __bf16* __restrict__ vT,
                                                   __bf16* __restrict__ ao) {
  extern __shared__ float S[];                 // [32][SROW] + rowinv[32]
  float* rowinv = S + 32 * SROW;

  int qb = blockIdx.x;                          // 0..63 (query block of 32)
  int bh = blockIdx.y;                          // 0..63
  int b = bh >> 4, h = bh & 15;
  int tid = threadIdx.x;
  int w = tid >> 5, lane = tid & 31, l15 = lane & 15, hi = lane >> 4;

  const __bf16* qbase = q  + ((size_t)bh * SEQ + qb * 32) * DHEAD;
  const __bf16* kbase = k  + (size_t)bh * SEQ * DHEAD;
  const __bf16* vbase = vT + (size_t)bh * DHEAD * SEQ;

  // Preload Q A-fragments: 2 row-blocks x 2 k-steps (d = 0..31, 32..63)
  v16bf aq[2][2];
#pragma unroll
  for (int i = 0; i < 2; ++i) {
    const __bf16* qr = qbase + (size_t)(i * 16 + l15) * DHEAD;
    aq[i][0] = load_a_frag(qr, 0, hi);
    aq[i][1] = load_a_frag(qr, 32, hi);
  }

  // ---- S = QK^T : wave w owns 256 consecutive key columns ----
  for (int jt = 0; jt < 16; ++jt) {
    int colbase = w * 256 + jt * 16;
    const __bf16* kr = kbase + (size_t)(colbase + l15) * DHEAD;
    v16bf fb0 = load_b_frag(kr, 0, hi);
    v16bf fb1 = load_b_frag(kr, 32, hi);
#pragma unroll
    for (int i = 0; i < 2; ++i) {
      v8f acc = {};
      acc = wmma_bf16(aq[i][0], fb0, acc);
      acc = wmma_bf16(aq[i][1], fb1, acc);
#pragma unroll
      for (int r = 0; r < 8; ++r)
        S[(size_t)(i * 16 + r + 8 * hi) * SROW + colbase + l15] = acc[r];
    }
  }
  __syncthreads();

  // ---- softmax + bf16 pack: 8 threads per row, 256-wide exclusive chunks ----
  {
    int row = tid >> 3, g = tid & 7;
    float* sr = S + (size_t)row * SROW + g * 256;
    float m = -3.0e38f;
    for (int t = 0; t < 64; ++t) {
      float4 f = reinterpret_cast<const float4*>(sr)[t];
      m = fmaxf(m, fmaxf(fmaxf(f.x, f.y), fmaxf(f.z, f.w)));
    }
#pragma unroll
    for (int d = 1; d < 8; d <<= 1) m = fmaxf(m, __shfl_xor(m, d, 8));

    // fused exp + pack-to-bf16 (overlay on own chunk; writes trail reads)
    float sum = 0.0f;
    __bf16* pb = reinterpret_cast<__bf16*>(sr);
    for (int blk = 0; blk < 16; ++blk) {
      float e[16];
#pragma unroll
      for (int t = 0; t < 4; ++t) {
        float4 f = reinterpret_cast<const float4*>(sr)[blk * 4 + t];
        e[t * 4 + 0] = __expf(f.x - m); e[t * 4 + 1] = __expf(f.y - m);
        e[t * 4 + 2] = __expf(f.z - m); e[t * 4 + 3] = __expf(f.w - m);
      }
      asm volatile("" ::: "memory");   // keep bf16 writes after fp32 reads
      union { __bf16 h[4]; ushort4 u; } o;
#pragma unroll
      for (int t = 0; t < 4; ++t) {
        sum += (e[t*4+0] + e[t*4+1]) + (e[t*4+2] + e[t*4+3]);
        o.h[0] = f2bf(e[t*4+0]); o.h[1] = f2bf(e[t*4+1]);
        o.h[2] = f2bf(e[t*4+2]); o.h[3] = f2bf(e[t*4+3]);
        reinterpret_cast<ushort4*>(pb)[blk * 4 + t] = o.u;
      }
    }
#pragma unroll
    for (int d = 1; d < 8; d <<= 1) sum += __shfl_xor(sum, d, 8);
    if (g == 0) rowinv[row] = 1.0f / sum;
  }
  __syncthreads();

  // ---- O = P V : wave w -> (row block i = w&1, d block j = w>>1) ----
  int i = w & 1, j = w >> 1;
  // bf16 P row base: row bytes = r*SROW*4; chunk g at +g*1024 bytes, col t at +t*2
  const __bf16* prow = reinterpret_cast<const __bf16*>(S) + (size_t)(i * 16 + l15) * (SROW * 2);
  const __bf16* vrow = vbase + (size_t)(j * 16 + l15) * SEQ;
  v8f acc = {};
  for (int ks = 0; ks < SEQ; ks += 64) {
    // batch-load 2 A (LDS) + 2 B (global) fragments, then 2 WMMA
    Frag16 a0, a1;
    {
      const __bf16* p0 = prow + ((ks      ) >> 8) * 512 + ((ks      ) & 255) + hi * 8;
      const __bf16* p1 = prow + ((ks + 32 ) >> 8) * 512 + ((ks + 32 ) & 255) + hi * 8;
      a0.u[0] = *reinterpret_cast<const uint4*>(p0);
      a0.u[1] = *reinterpret_cast<const uint4*>(p0 + 16);
      a1.u[0] = *reinterpret_cast<const uint4*>(p1);
      a1.u[1] = *reinterpret_cast<const uint4*>(p1 + 16);
    }
    v16bf fb0 = load_b_frag(vrow, ks, hi);
    v16bf fb1 = load_b_frag(vrow, ks + 32, hi);
    acc = wmma_bf16(a0.v, fb0, acc);
    acc = wmma_bf16(a1.v, fb1, acc);
  }
  // Epilogue: apply 1/rowsum, write bf16 to attn-out [b, n, h*64 + d]
#pragma unroll
  for (int r = 0; r < 8; ++r) {
    int rr = i * 16 + r + 8 * hi;
    int n  = qb * 32 + rr;
    int col = h * DHEAD + j * 16 + l15;
    float o = acc[r] * rowinv[rr];
    ao[((size_t)(b * SEQ + n)) * DMODEL + col] = f2bf(o);
  }
}

// ---------------------------------------------------------------------------
// 5) Output GEMM: [8192 x 1024] x [1024 x 1024] + bias -> fp32 out; 32x64 tile
// ---------------------------------------------------------------------------
__global__ __launch_bounds__(256) void out_gemm_kernel(const __bf16* __restrict__ ao,
                                                       const __bf16* __restrict__ wT,
                                                       const float* __restrict__ bias,
                                                       float* __restrict__ out) {
  int wave = blockIdx.x * (blockDim.x >> 5) + (threadIdx.x >> 5);
  int lane = threadIdx.x & 31, l15 = lane & 15, hi = lane >> 4;
  const int MT = ROWS / 32;                    // 256
  int tm = wave % MT;
  int tn = wave / MT;                          // 0..15
  const __bf16* a0 = ao + (size_t)(tm * 32 + l15) * DMODEL;
  const __bf16* a1 = a0 + 16 * DMODEL;
  const __bf16* b0 = wT + (size_t)(tn * 64 + l15) * DMODEL;

  v8f acc[2][4] = {};
  for (int kk = 0; kk < DMODEL; kk += 32) {
    __builtin_prefetch(a0 + kk + 256, 0, 3);
    __builtin_prefetch(b0 + kk + 256, 0, 3);
    v16bf fa0 = load_a_frag(a0, kk, hi);
    v16bf fa1 = load_a_frag(a1, kk, hi);
    v16bf fb[4];
#pragma unroll
    for (int j = 0; j < 4; ++j)
      fb[j] = load_b_frag(b0 + (size_t)j * 16 * DMODEL, kk, hi);
#pragma unroll
    for (int j = 0; j < 4; ++j) {
      acc[0][j] = wmma_bf16(fa0, fb[j], acc[0][j]);
      acc[1][j] = wmma_bf16(fa1, fb[j], acc[1][j]);
    }
  }
#pragma unroll
  for (int i = 0; i < 2; ++i) {
#pragma unroll
    for (int j = 0; j < 4; ++j) {
      int N = tn * 64 + j * 16 + l15;
      float bb = bias[N];
#pragma unroll
      for (int r = 0; r < 8; ++r) {
        int M = tm * 32 + i * 16 + r + 8 * hi;
        out[(size_t)M * DMODEL + N] = acc[i][j][r] + bb;
      }
    }
  }
}

// ---------------------------------------------------------------------------
// Launcher
// ---------------------------------------------------------------------------
extern "C" void kernel_launch(void* const* d_in, const int* in_sizes, int n_in,
                              void* d_out, int out_size, void* d_ws, size_t ws_size,
                              hipStream_t stream) {
  (void)in_sizes; (void)n_in; (void)out_size; (void)ws_size;
  const float* x     = (const float*)d_in[0];
  const float* w_qkv = (const float*)d_in[1];
  const float* w_out = (const float*)d_in[2];
  const float* b_out = (const float*)d_in[3];
  float* out = (float*)d_out;

  char* ws = (char*)d_ws;
  __bf16* xb     = (__bf16*)(ws);                                  // 16 MB
  __bf16* wqkvT  = (__bf16*)(ws + (size_t)16777216);               //  6 MB
  __bf16* woutT  = (__bf16*)(ws + (size_t)23068672);               //  2 MB
  __bf16* qbuf   = (__bf16*)(ws + (size_t)25165824);               // 16 MB
  __bf16* kbuf   = (__bf16*)(ws + (size_t)41943040);               // 16 MB
  __bf16* vTbuf  = (__bf16*)(ws + (size_t)58720256);               // 16 MB
  __bf16* aobuf  = (__bf16*)(ws + (size_t)75497472);               // 16 MB  (total 88 MB)

  // 1) casts
  {
    int n4 = ROWS * DMODEL / 4;                                    // 2,097,152
    cast4_kernel<<<n4 / 256, 256, 0, stream>>>(x, xb, n4);
  }
  {
    int tot = DMODEL * N_QKV;                                      // 3,145,728
    cast_tr_kernel<<<tot / 256, 256, 0, stream>>>(w_qkv, wqkvT, DMODEL, N_QKV);
  }
  {
    int tot = DMODEL * DMODEL;                                     // 1,048,576
    cast_tr_kernel<<<tot / 256, 256, 0, stream>>>(w_out, woutT, DMODEL, DMODEL);
  }

  // 2) QKV projection: (8192/32)*(3072/64) = 12288 wave tiles / 8 per block
  qkv_gemm_kernel<<<(ROWS / 32) * (N_QKV / 64) / 8, 256, 0, stream>>>(xb, wqkvT, qbuf, kbuf, vTbuf);

  // 3) attention: grid (64 q-blocks, 64 batch*heads), 262784 B dynamic LDS
  {
    dim3 grid(SEQ / 32, BATCH * NHEAD);
    size_t shmem = (size_t)(32 * SROW) * sizeof(float) + 32 * sizeof(float);
    attn_kernel<<<grid, 256, shmem, stream>>>(qbuf, kbuf, vTbuf, aobuf);
  }

  // 4) output projection + bias: (8192/32)*(1024/64) = 4096 wave tiles / 8 per block
  out_gemm_kernel<<<(ROWS / 32) * (DMODEL / 64) / 8, 256, 0, stream>>>(aobuf, woutT, b_out, out);
}
